// CausalSelfAttention_15607911153972
// MI455X (gfx1250) — matrix operand order repaired and checked by reading, not compiled, not run on hardware
//
#include <hip/hip_runtime.h>

// ---------------- CDNA5 WMMA types ----------------
typedef __attribute__((ext_vector_type(16))) __bf16 v16bf;
typedef __attribute__((ext_vector_type(8)))  float  v8f;
typedef __attribute__((ext_vector_type(4)))  int    v4i;

union ABf { v16bf v; unsigned int u[8]; unsigned short s[16]; };

// AS-qualified pointer types for the async builtin: (AS1 v4i*, AS3 v4i*, imm, imm)
typedef __attribute__((address_space(1))) v4i as1_v4i;
typedef __attribute__((address_space(3))) v4i as3_v4i;

__device__ __forceinline__ unsigned short f32_to_bf16(float f) {
    unsigned int x = __float_as_uint(f);
    unsigned int r = x + 0x7FFFu + ((x >> 16) & 1u);   // RNE
    return (unsigned short)(r >> 16);
}

__device__ __forceinline__ v8f v8f_zero() {
    v8f r;
#pragma unroll
    for (int i = 0; i < 8; ++i) r[i] = 0.0f;
    return r;
}

// ---- async global->LDS B128 copy (ASYNCcnt path) ----
__device__ __forceinline__ void async_copy_b128(const unsigned short* g, unsigned short* l) {
#if __has_builtin(__builtin_amdgcn_global_load_async_to_lds_b128)
    __builtin_amdgcn_global_load_async_to_lds_b128((as1_v4i*)(void*)g, (as3_v4i*)(void*)l, 0, 0);
#else
    unsigned int lds = (unsigned int)(unsigned long long)l;
    unsigned long long ga = (unsigned long long)g;
    asm volatile("global_load_async_to_lds_b128 %0, %1, off" :: "v"(lds), "v"(ga) : "memory");
#endif
}

__device__ __forceinline__ void wait_async0() {
#if __has_builtin(__builtin_amdgcn_s_wait_asynccnt)
    __builtin_amdgcn_s_wait_asynccnt(0);
#else
    asm volatile("s_wait_asynccnt 0x0" ::: "memory");
#endif
}

// A-matrix (bf16 16x32, MxK) fragment load from row-major src (element (m,k) = src[m*ld+k]).
// Contiguous k order used here (B uses the same): lane L: m=L&15, half=L>>4; VGPR v0..7 hold K=half*16+2v,+1
__device__ __forceinline__ void load_a16x32(const unsigned short* src, int ld, ABf& a) {
    int lane = threadIdx.x & 31;
    int m = lane & 15, h = lane >> 4;
    const unsigned short* row = src + m * ld;
#pragma unroll
    for (int v = 0; v < 8; ++v) a.u[v] = *(const unsigned int*)(row + h * 16 + 2 * v);
}

// B-matrix (bf16 32x16, KxN) fragment where B = M^T and M is row-major (element B(k,n)=M[n*ld+k]).
// Contiguous k order used here (same as A): lane L: n=L&15, khalf=L>>4; VGPR v holds K=khalf*16+2v,+1 -> contiguous pair in M's row n.
__device__ __forceinline__ void load_bt16x32(const unsigned short* src, int ld, ABf& b) {
    int lane = threadIdx.x & 31;
    int n = lane & 15, kh = lane >> 4;
    const unsigned short* row = src + n * ld + kh * 16;
#pragma unroll
    for (int v = 0; v < 8; ++v) b.u[v] = *(const unsigned int*)(row + 2 * v);
}

__device__ __forceinline__ v8f wmma_bf16(const ABf& a, const ABf& b, v8f c) {
    return __builtin_amdgcn_wmma_f32_16x16x32_bf16(false, a.v, false, b.v, (short)0, c, false, false);
}

// ---------------- problem constants ----------------
#define BATCH 4
#define SEQ   2048
#define CDIM  1024
#define NH    16
#define HD    64
#define MROWS (BATCH * SEQ)   // 8192
#define LDA_S 40              // A tile LDS pitch (u16): 80B rows -> 16B-aligned B128 chunks
#define LDB_S 34              // B tile LDS pitch (u16)

// ---------------- fp32 -> bf16 convert ----------------
__global__ __launch_bounds__(256) void cvt_f32_bf16(const float* __restrict__ in,
                                                    unsigned short* __restrict__ out, int n) {
    int i = blockIdx.x * 256 + threadIdx.x;
    int stride = gridDim.x * 256;
    for (; i < n; i += stride) out[i] = f32_to_bf16(in[i]);
}

// ---------------- shared GEMM pipeline pieces ----------------
struct BReg { unsigned int d[8]; };

// Phase 1: issue async A-tile copy into sa(next) and B-tile global loads into registers.
__device__ __forceinline__ void stage_issue(const unsigned short* __restrict__ A,
                                            const unsigned short* __restrict__ Bw, int Nn,
                                            unsigned short* sa, int m0, int n0, int k0, int t,
                                            BReg& br) {
    // A: 128x32 bf16 = 512 x 16B chunks; 256 threads x 2 async B128 copies
#pragma unroll
    for (int i = 0; i < 2; ++i) {
        int c = t + 256 * i;
        int row = c >> 2, seg = c & 3;
        async_copy_b128(A + (m0 + row) * CDIM + k0 + seg * 8, sa + row * LDA_S + seg * 8);
    }
    // B: 32x128 -> registers (committed transposed later, after compute reads)
#pragma unroll
    for (int i = 0; i < 8; ++i) {
        int lin = t + 256 * i;
        int row = lin >> 6, c2 = lin & 63;
        br.d[i] = *(const unsigned int*)(Bw + (k0 + row) * Nn + n0 + c2 * 2);
    }
    // prefetch two tiles ahead (global_prefetch_b8)
    if (k0 + 64 < CDIM) {
        __builtin_prefetch(A + (m0 + (t >> 1)) * CDIM + k0 + 64 + (t & 1) * 16, 0, 3);
        __builtin_prefetch(Bw + (k0 + 64 + (t >> 3)) * Nn + n0 + (t & 7) * 16, 0, 3);
    }
}

// Phase 3: commit B registers to LDS transposed [n][k]
__device__ __forceinline__ void stage_commit(unsigned short* sb, int t, const BReg& br) {
#pragma unroll
    for (int i = 0; i < 8; ++i) {
        int lin = t + 256 * i;
        int row = lin >> 6, c2 = lin & 63;
        int n = c2 * 2;
        sb[n * LDB_S + row]       = (unsigned short)(br.d[i] & 0xffffu);
        sb[(n + 1) * LDB_S + row] = (unsigned short)(br.d[i] >> 16);
    }
}

// Phase 2: fragment loads + 8 WMMAs from current buffers
__device__ __forceinline__ void compute_tile(const unsigned short* sa, const unsigned short* sb,
                                             int wm, int wn, v8f acc[4][2]) {
    ABf af[4];
#pragma unroll
    for (int i = 0; i < 4; ++i) load_a16x32(&sa[(wm * 64 + i * 16) * LDA_S], LDA_S, af[i]);
    ABf bfr[2];
#pragma unroll
    for (int j = 0; j < 2; ++j) load_bt16x32(&sb[(wn * 32 + j * 16) * LDB_S], LDB_S, bfr[j]);
#pragma unroll
    for (int i = 0; i < 4; ++i)
#pragma unroll
        for (int j = 0; j < 2; ++j) acc[i][j] = wmma_bf16(af[i], bfr[j], acc[i][j]);
}

// ---------------- GEMM1: qkv = x @ w_attn + b_attn, scatter to Q,K (BHTD) and Vt (BHDT) ----------------
__global__ __launch_bounds__(256) void gemm_qkv(const unsigned short* __restrict__ A,   // [8192,1024] bf16
                                                const unsigned short* __restrict__ Bw,  // [1024,3072] bf16
                                                const float* __restrict__ bias,         // [3072]
                                                unsigned short* __restrict__ Qo,
                                                unsigned short* __restrict__ Ko,
                                                unsigned short* __restrict__ Vto) {
    const int N = 3 * CDIM;
    __shared__ __align__(16) unsigned short sa[2][128 * LDA_S];
    __shared__ __align__(16) unsigned short sb[2][128 * LDB_S];
    int t = threadIdx.x;
    int lane = t & 31, w = t >> 5;
    int wm = w >> 2, wn = w & 3;
    int hlf = lane >> 4, nl = lane & 15;
    int m0 = blockIdx.y * 128, n0 = blockIdx.x * 128;

    v8f acc[4][2];
#pragma unroll
    for (int i = 0; i < 4; ++i)
#pragma unroll
        for (int j = 0; j < 2; ++j) acc[i][j] = v8f_zero();

    BReg br;
    stage_issue(A, Bw, N, sa[0], m0, n0, 0, t, br);
    stage_commit(sb[0], t, br);
    wait_async0();
    __syncthreads();

    const int KT = CDIM / 32;
    for (int kt = 0; kt < KT; ++kt) {
        int cur = kt & 1;
        bool has_next = (kt + 1 < KT);
        if (has_next) stage_issue(A, Bw, N, sa[1 - cur], m0, n0, (kt + 1) * 32, t, br);
        compute_tile(sa[cur], sb[cur], wm, wn, acc);
        if (has_next) stage_commit(sb[1 - cur], t, br);
        wait_async0();
        __syncthreads();
    }

    // epilogue: add bias, convert, scatter into Q / K / Vt
#pragma unroll
    for (int i = 0; i < 4; ++i) {
        int mbase = m0 + wm * 64 + i * 16;
#pragma unroll
        for (int j = 0; j < 2; ++j) {
            int nbase = n0 + wn * 32 + j * 16;
            int which = nbase >> 10;          // 0=q,1=k,2=v (uniform per tile)
            int c = nbase & 1023;
            int hh = c >> 6;                  // head (uniform per tile)
            int dbase = c & 63;
            int d = dbase + nl;
            float bv = bias[nbase + nl];
#pragma unroll
            for (int r = 0; r < 8; ++r) {
                int mglob = mbase + r + 8 * hlf;
                int bb = mglob >> 11, tt = mglob & 2047;
                unsigned short val = f32_to_bf16(acc[i][j][r] + bv);
                if (which == 0)
                    Qo[(((bb * NH + hh) * SEQ) + tt) * HD + d] = val;
                else if (which == 1)
                    Ko[(((bb * NH + hh) * SEQ) + tt) * HD + d] = val;
                else
                    Vto[(((bb * NH + hh) * HD) + d) * SEQ + tt] = val;
            }
        }
    }
}

// ---------------- Flash attention (causal, online softmax) ----------------
// one wave = 16 query rows; block = 8 waves = 128 query rows for one (b,h)
__global__ __launch_bounds__(256) void flash_attn(const unsigned short* __restrict__ Q,   // [B,H,T,D]
                                                  const unsigned short* __restrict__ Kb,  // [B,H,T,D]
                                                  const unsigned short* __restrict__ Vt,  // [B,H,D,T]
                                                  unsigned short* __restrict__ Aout) {    // [B,T,C]
    __shared__ __align__(16) unsigned short pbuf[8][16 * 34];
    int t = threadIdx.x, lane = t & 31, w = t >> 5;
    int h = blockIdx.y, b = blockIdx.z;
    int q0 = blockIdx.x * 128 + w * 16;
    int hlf = lane >> 4, nl = lane & 15;

    const unsigned short* Qh = Q  + ((size_t)(b * NH + h) * SEQ) * HD;
    const unsigned short* Kh = Kb + ((size_t)(b * NH + h) * SEQ) * HD;
    const unsigned short* Vh = Vt + ((size_t)(b * NH + h) * HD) * SEQ;

    ABf qa0, qa1;
    load_a16x32(Qh + q0 * HD, HD, qa0);
    load_a16x32(Qh + q0 * HD + 32, HD, qa1);

    v8f o0 = v8f_zero(), o1 = v8f_zero(), o2 = v8f_zero(), o3 = v8f_zero();
    float mrow[8], lrow[8];
#pragma unroll
    for (int r = 0; r < 8; ++r) { mrow[r] = -1e30f; lrow[r] = 0.0f; }

    unsigned short* pb = &pbuf[w][0];
    const float scale = 0.125f;   // 1/sqrt(64)
    int ktmax = (q0 + 15) >> 5;

    for (int kt = 0; kt <= ktmax; ++kt) {
        int kbase = kt * 32;
        v8f s[2];
#pragma unroll
        for (int j = 0; j < 2; ++j) {
            ABf b0, b1;
            load_bt16x32(Kh + (kbase + j * 16) * HD, HD, b0);       // (K tile)^T, d=0..31
            load_bt16x32(Kh + (kbase + j * 16) * HD + 32, HD, b1);  // d=32..63
            v8f a = v8f_zero();
            a = wmma_bf16(qa0, b0, a);
            a = wmma_bf16(qa1, b1, a);
            s[j] = a * scale;
        }
        if (kbase + 31 > q0) {                       // diagonal block: causal mask
#pragma unroll
            for (int j = 0; j < 2; ++j) {
                int kidx = kbase + j * 16 + nl;
#pragma unroll
                for (int r = 0; r < 8; ++r) {
                    int qidx = q0 + r + 8 * hlf;
                    if (kidx > qidx) s[j][r] = -1e30f;
                }
            }
        }
        // online softmax: row max / exp / row sum (rows live across 16 lanes within a half)
        float newm[8], rsum[8];
#pragma unroll
        for (int r = 0; r < 8; ++r) {
            float v = fmaxf(s[0][r], s[1][r]);
#pragma unroll
            for (int msk = 8; msk >= 1; msk >>= 1) v = fmaxf(v, __shfl_xor(v, msk, 32));
            newm[r] = fmaxf(mrow[r], v);
        }
#pragma unroll
        for (int r = 0; r < 8; ++r) {
            float p0 = __expf(s[0][r] - newm[r]);
            float p1 = __expf(s[1][r] - newm[r]);
            s[0][r] = p0; s[1][r] = p1;
            float v = p0 + p1;
#pragma unroll
            for (int msk = 8; msk >= 1; msk >>= 1) v += __shfl_xor(v, msk, 32);
            rsum[r] = v;
        }
#pragma unroll
        for (int r = 0; r < 8; ++r) {
            float corr = __expf(mrow[r] - newm[r]);
            lrow[r] = lrow[r] * corr + rsum[r];
            mrow[r] = newm[r];
            o0[r] *= corr; o1[r] *= corr; o2[r] *= corr; o3[r] *= corr;
        }
        // P (C layout f32) -> LDS bf16 row-major -> reload in A layout
#pragma unroll
        for (int j = 0; j < 2; ++j)
#pragma unroll
            for (int r = 0; r < 8; ++r)
                pb[(r + 8 * hlf) * 34 + j * 16 + nl] = f32_to_bf16(s[j][r]);
        asm volatile("s_wait_dscnt 0" ::: "memory");   // wave-private LDS RAW
        ABf pa;
        load_a16x32(pb, 34, pa);
        // O += P @ V : B fragments come contiguous from pre-transposed V [d][t]
#pragma unroll
        for (int dt = 0; dt < 4; ++dt) {
            ABf vb;
            load_bt16x32(Vh + (dt * 16) * SEQ + kbase, SEQ, vb);
            v8f* op = (dt == 0) ? &o0 : (dt == 1) ? &o1 : (dt == 2) ? &o2 : &o3;
            *op = wmma_bf16(pa, vb, *op);
        }
    }

    // finalize: O / l, write bf16 to [b, t, h*64+d]
#pragma unroll
    for (int r = 0; r < 8; ++r) {
        float inv = 1.0f / lrow[r];
        int tq = q0 + r + 8 * hlf;
        unsigned short* orow = Aout + ((size_t)(b * SEQ + tq)) * CDIM + h * HD;
        orow[0 * 16 + nl] = f32_to_bf16(o0[r] * inv);
        orow[1 * 16 + nl] = f32_to_bf16(o1[r] * inv);
        orow[2 * 16 + nl] = f32_to_bf16(o2[r] * inv);
        orow[3 * 16 + nl] = f32_to_bf16(o3[r] * inv);
    }
}

// ---------------- GEMM2: out = attn_out @ w_proj + b_proj (fp32 out) ----------------
__global__ __launch_bounds__(256) void gemm_proj(const unsigned short* __restrict__ A,   // [8192,1024] bf16
                                                 const unsigned short* __restrict__ Bw,  // [1024,1024] bf16
                                                 const float* __restrict__ bias,         // [1024]
                                                 float* __restrict__ out) {              // [8192,1024] f32
    const int N = CDIM;
    __shared__ __align__(16) unsigned short sa[2][128 * LDA_S];
    __shared__ __align__(16) unsigned short sb[2][128 * LDB_S];
    int t = threadIdx.x;
    int lane = t & 31, w = t >> 5;
    int wm = w >> 2, wn = w & 3;
    int hlf = lane >> 4, nl = lane & 15;
    int m0 = blockIdx.y * 128, n0 = blockIdx.x * 128;

    v8f acc[4][2];
#pragma unroll
    for (int i = 0; i < 4; ++i)
#pragma unroll
        for (int j = 0; j < 2; ++j) acc[i][j] = v8f_zero();

    BReg br;
    stage_issue(A, Bw, N, sa[0], m0, n0, 0, t, br);
    stage_commit(sb[0], t, br);
    wait_async0();
    __syncthreads();

    const int KT = CDIM / 32;
    for (int kt = 0; kt < KT; ++kt) {
        int cur = kt & 1;
        bool has_next = (kt + 1 < KT);
        if (has_next) stage_issue(A, Bw, N, sa[1 - cur], m0, n0, (kt + 1) * 32, t, br);
        compute_tile(sa[cur], sb[cur], wm, wn, acc);
        if (has_next) stage_commit(sb[1 - cur], t, br);
        wait_async0();
        __syncthreads();
    }

#pragma unroll
    for (int i = 0; i < 4; ++i) {
        int mbase = m0 + wm * 64 + i * 16;
#pragma unroll
        for (int j = 0; j < 2; ++j) {
            int nbase = n0 + wn * 32 + j * 16;
            float bv = bias[nbase + nl];
#pragma unroll
            for (int r = 0; r < 8; ++r) {
                int mglob = mbase + r + 8 * hlf;
                out[(size_t)mglob * CDIM + nbase + nl] = acc[i][j][r] + bv;
            }
        }
    }
}

// ---------------- launcher ----------------
extern "C" void kernel_launch(void* const* d_in, const int* in_sizes, int n_in,
                              void* d_out, int out_size, void* d_ws, size_t ws_size,
                              hipStream_t stream) {
    const float* x      = (const float*)d_in[0];   // [4,2048,1024]
    const float* w_attn = (const float*)d_in[1];   // [1024,3072]
    const float* b_attn = (const float*)d_in[2];   // [3072]
    const float* w_proj = (const float*)d_in[3];   // [1024,1024]
    const float* b_proj = (const float*)d_in[4];   // [1024]
    float* out = (float*)d_out;

    const int NX  = MROWS * CDIM;           // 8,388,608
    const int NWA = CDIM * 3 * CDIM;        // 3,145,728
    const int NWP = CDIM * CDIM;            // 1,048,576
    const int NQ  = BATCH * NH * SEQ * HD;  // 8,388,608

    unsigned short* ws  = (unsigned short*)d_ws;
    unsigned short* xb  = ws;
    unsigned short* wab = xb + NX;
    unsigned short* wpb = wab + NWA;
    unsigned short* Qb  = wpb + NWP;
    unsigned short* Kb  = Qb + NQ;
    unsigned short* Vtb = Kb + NQ;
    unsigned short* Ab  = Vtb + NQ;

    cvt_f32_bf16<<<4096, 256, 0, stream>>>(x, xb, NX);
    cvt_f32_bf16<<<2048, 256, 0, stream>>>(w_attn, wab, NWA);
    cvt_f32_bf16<<<1024, 256, 0, stream>>>(w_proj, wpb, NWP);

    gemm_qkv<<<dim3(24, 64), 256, 0, stream>>>(xb, wab, b_attn, Qb, Kb, Vtb);
    flash_attn<<<dim3(SEQ / 128, NH, BATCH), 256, 0, stream>>>(Qb, Kb, Vtb, Ab);
    gemm_proj<<<dim3(8, 64), 256, 0, stream>>>(Ab, wpb, b_proj, out);
}
